// Bilinear_44083544326835
// MI455X (gfx1250) — matrix operand order repaired and checked by reading, not compiled, not run on hardware
//
#include <hip/hip_runtime.h>

typedef __attribute__((ext_vector_type(16))) __bf16 v16bf;
typedef __attribute__((ext_vector_type(8)))  float  v8f;

#define B_SZ   256
#define U_SZ   512
#define D0_SZ  512
#define D1_SZ  1024
#define JB     64                  // j-slice width per workgroup
#define JP     (JB + 8)            // padded LDS row stride (144B) -> conflict-free quads
#define NSPLITJ (D1_SZ / JB)       // 16 j-slices
#define IB     4                   // i-rows staged per barrier round

// f32 -> bf16 RNE, packed pair; high halves gathered with one v_perm_b32.
__device__ __forceinline__ unsigned int pk2bf(float a, float b) {
    unsigned int ua = __float_as_uint(a);
    unsigned int ub = __float_as_uint(b);
    ua += 0x7FFFu + ((ua >> 16) & 1u);
    ub += 0x7FFFu + ((ub >> 16) & 1u);
    return __builtin_amdgcn_perm(ub, ua, 0x07060302u);
}

__global__ __launch_bounds__(256) void bilinear_bias_init(const float* __restrict__ bias,
                                                          float* __restrict__ out) {
    int idx = blockIdx.x * 256 + threadIdx.x;     // out is (B, U) row-major
    out[idx] = bias[idx & (U_SZ - 1)];
}

__global__ __launch_bounds__(256) void bilinear_wmma(const float* __restrict__ x,
                                                     const float* __restrict__ y,
                                                     const float* __restrict__ w,
                                                     float* __restrict__ out) {
    __shared__ __align__(16) unsigned short Yb[B_SZ][JP];      // 36 KB: A rows (b x j), bf16
    __shared__ __align__(16) unsigned short Wb[IB][16][JP];    //  9 KB: B tiles (u x j), bf16
    __shared__ __align__(16) float          Xs[IB][B_SZ];      //  4 KB: x[:, i..i+3]

    const int u0 = blockIdx.x * 16;        // which 16 u's
    const int j0 = blockIdx.y * JB;        // which j slice

    const int t    = threadIdx.x;
    const int lane = t & 31;
    const int wv   = t >> 5;               // wave 0..7
    const int half = lane >> 4;            // lane half (WMMA layout)
    const int l16  = lane & 15;
    const int mb0  = (wv * 2 + 0) * 16;    // each wave owns two 16-row M tiles
    const int mb1  = (wv * 2 + 1) * 16;

    // ---- one-time: stage bf16 y slice; thread t owns row b == t
    {
        const float4* yrow = (const float4*)(y + (size_t)t * D1_SZ + j0);
        #pragma unroll
        for (int q = 0; q < JB / 4; ++q) {
            float4 v = yrow[q];
            uint2 p;
            p.x = pk2bf(v.x, v.y);
            p.y = pk2bf(v.z, v.w);
            *(uint2*)&Yb[t][q * 4] = p;
        }
    }
    __syncthreads();

    // ---- hoist A fragments (independent of i): ISA 16-bit A 16x32 layout
    v16bf a00, a01, a10, a11;
    ((uint4*)&a00)[0] = *(const uint4*)&Yb[mb0 + l16][ 0 + half * 8];
    ((uint4*)&a00)[1] = *(const uint4*)&Yb[mb0 + l16][16 + half * 8];
    ((uint4*)&a01)[0] = *(const uint4*)&Yb[mb0 + l16][32 + half * 8];
    ((uint4*)&a01)[1] = *(const uint4*)&Yb[mb0 + l16][48 + half * 8];
    ((uint4*)&a10)[0] = *(const uint4*)&Yb[mb1 + l16][ 0 + half * 8];
    ((uint4*)&a10)[1] = *(const uint4*)&Yb[mb1 + l16][16 + half * 8];
    ((uint4*)&a11)[0] = *(const uint4*)&Yb[mb1 + l16][32 + half * 8];
    ((uint4*)&a11)[1] = *(const uint4*)&Yb[mb1 + l16][48 + half * 8];

    v8f acc0 = {}; v8f acc1 = {};

    // staging assignment for w: 16 threads per u-row, 4 floats each
    const int uu = t >> 4;                 // 0..15
    const int kk = (t & 15) * 4;           // 0..60
    const float* wbase = w + (size_t)(u0 + uu) * (D0_SZ * D1_SZ) + j0 + kk;

    for (int i = 0; i < D0_SZ; i += IB) {
        // stage bf16 w tiles for i..i+3, plus the x column quad
        {
            #pragma unroll
            for (int s = 0; s < IB; ++s) {
                float4 v = *(const float4*)(wbase + (size_t)(i + s) * D1_SZ);
                uint2 q;
                q.x = pk2bf(v.x, v.y);
                q.y = pk2bf(v.z, v.w);
                *(uint2*)&Wb[s][uu][kk] = q;
            }
            float4 xv = *(const float4*)(x + (size_t)t * D0_SZ + i);
            Xs[0][t] = xv.x;
            Xs[1][t] = xv.y;
            Xs[2][t] = xv.z;
            Xs[3][t] = xv.w;
        }
        if (i + 2 * IB < D0_SZ) {
            #pragma unroll
            for (int s = 0; s < IB; ++s)
                __builtin_prefetch(wbase + (size_t)(i + IB + s) * D1_SZ, 0, 3);
        }
        __syncthreads();

        #pragma unroll
        for (int s = 0; s < IB; ++s) {
            // B fragments: lane column n=l16, K = 8*half+s (s<8), 16+8*half+(s-8) (s>=8)
            v16bf b0, b1;
            ((uint4*)&b0)[0] = *(const uint4*)&Wb[s][l16][ 0 + half * 8];
            ((uint4*)&b0)[1] = *(const uint4*)&Wb[s][l16][16 + half * 8];
            ((uint4*)&b1)[0] = *(const uint4*)&Wb[s][l16][32 + half * 8];
            ((uint4*)&b1)[1] = *(const uint4*)&Wb[s][l16][48 + half * 8];

            // P_i = y_slice * w[:, i+s, slice]^T  (K = 64)
            v8f p0 = {}; v8f p1 = {};
            p0 = __builtin_amdgcn_wmma_f32_16x16x32_bf16(false, a00, false, b0, (short)0, p0, false, false);
            p0 = __builtin_amdgcn_wmma_f32_16x16x32_bf16(false, a01, false, b1, (short)0, p0, false, false);
            p1 = __builtin_amdgcn_wmma_f32_16x16x32_bf16(false, a10, false, b0, (short)0, p1, false, false);
            p1 = __builtin_amdgcn_wmma_f32_16x16x32_bf16(false, a11, false, b1, (short)0, p1, false, false);

            // fold x[:, i+s] scaling into the accumulation (C/D row m = r + 8*half)
            float4 xv0a = *(const float4*)&Xs[s][mb0 + half * 8];
            float4 xv0b = *(const float4*)&Xs[s][mb0 + half * 8 + 4];
            float4 xv1a = *(const float4*)&Xs[s][mb1 + half * 8];
            float4 xv1b = *(const float4*)&Xs[s][mb1 + half * 8 + 4];
            acc0[0] += xv0a.x * p0[0]; acc0[1] += xv0a.y * p0[1];
            acc0[2] += xv0a.z * p0[2]; acc0[3] += xv0a.w * p0[3];
            acc0[4] += xv0b.x * p0[4]; acc0[5] += xv0b.y * p0[5];
            acc0[6] += xv0b.z * p0[6]; acc0[7] += xv0b.w * p0[7];
            acc1[0] += xv1a.x * p1[0]; acc1[1] += xv1a.y * p1[1];
            acc1[2] += xv1a.z * p1[2]; acc1[3] += xv1a.w * p1[3];
            acc1[4] += xv1b.x * p1[4]; acc1[5] += xv1b.y * p1[5];
            acc1[6] += xv1b.z * p1[6]; acc1[7] += xv1b.w * p1[7];
        }
        __syncthreads();
    }

    // combine j-slice partials: hardware f32 atomics at agent scope
    const int ucol = u0 + l16;
    #pragma unroll
    for (int r = 0; r < 8; ++r) {
        __hip_atomic_fetch_add(&out[(mb0 + r + half * 8) * U_SZ + ucol], acc0[r],
                               __ATOMIC_RELAXED, __HIP_MEMORY_SCOPE_AGENT);
        __hip_atomic_fetch_add(&out[(mb1 + r + half * 8) * U_SZ + ucol], acc1[r],
                               __ATOMIC_RELAXED, __HIP_MEMORY_SCOPE_AGENT);
    }
}

extern "C" void kernel_launch(void* const* d_in, const int* in_sizes, int n_in,
                              void* d_out, int out_size, void* d_ws, size_t ws_size,
                              hipStream_t stream) {
    const float* x    = (const float*)d_in[0];
    const float* y    = (const float*)d_in[1];
    const float* w    = (const float*)d_in[2];
    const float* bias = (const float*)d_in[3];
    float* out = (float*)d_out;

    bilinear_bias_init<<<dim3((B_SZ * U_SZ) / 256), dim3(256), 0, stream>>>(bias, out);
    bilinear_wmma<<<dim3(U_SZ / 16, NSPLITJ), dim3(256), 0, stream>>>(x, y, w, out);
}
